// GatedLSTM_73323681677956
// MI455X (gfx1250) — matrix operand order repaired and checked by reading, not compiled, not run on hardware
//
#include <hip/hip_runtime.h>

typedef __bf16 bf16_t;
typedef __attribute__((ext_vector_type(4)))  __bf16 v4bf;
typedef __attribute__((ext_vector_type(8)))  __bf16 v8bf;
typedef __attribute__((ext_vector_type(16))) __bf16 v16bf;
typedef __attribute__((ext_vector_type(8)))  float  v8f;

constexpr int B_  = 64;
constexpr int T_  = 256;
constexpr int IN_ = 512;
constexpr int H_  = 1024;
constexpr int BT_ = B_ * T_;   // 16384

// fast sigmoid/tanh: v_exp_f32 + v_rcp_f32 (bf16 matmul noise >> 1-ulp rcp error)
__device__ __forceinline__ float sigm(float x) {
    return __builtin_amdgcn_rcpf(1.0f + __expf(-x));
}
__device__ __forceinline__ float tanh_fast(float x) {
    return __builtin_fmaf(2.0f, sigm(2.0f * x), -1.0f);
}

// A-fragment (16x32 bf16), lane<16: row m0+lane, K={0..15}; lane>=16: K={16..31}
__device__ __forceinline__ v16bf load_a_frag(const bf16_t* __restrict__ base, int ld,
                                             int m0, int k0, int lane) {
    const bf16_t* p = base + (size_t)(m0 + (lane & 15)) * ld + k0 + ((lane & 16) ? 16 : 0);
    v8bf lo = *(const v8bf*)p;
    v8bf hi = *(const v8bf*)(p + 8);
    return __builtin_shufflevector(lo, hi, 0,1,2,3,4,5,6,7,8,9,10,11,12,13,14,15);
}

// B-fragment (32x16 bf16), column n = row n of U (we need U^T); lane<16: K=k0..k0+15, else +16
__device__ __forceinline__ v16bf load_b_frag(const bf16_t* __restrict__ base, int ld,
                                             int n0, int k0, int lane) {
    const bf16_t* p = base + (size_t)(n0 + (lane & 15)) * ld + k0 + ((lane & 16) ? 16 : 0);
    return *(const v16bf*)p;
}

__device__ __forceinline__ v8f wmma_bf16(v16bf a, v16bf b, v8f c) {
    return __builtin_amdgcn_wmma_f32_16x16x32_bf16(false, a, false, b, (short)0, c, false, false);
}

// ---------------- conversion / init ----------------

__global__ __launch_bounds__(256) void f32_to_bf16_kernel(const float* __restrict__ src,
                                                          bf16_t* __restrict__ dst, int n) {
    int i = (blockIdx.x * blockDim.x + threadIdx.x) * 4;
    if (i < n) {
        float4 v = *(const float4*)(src + i);
        v4bf o = { (bf16_t)v.x, (bf16_t)v.y, (bf16_t)v.z, (bf16_t)v.w };
        *(v4bf*)(dst + i) = o;
    }
}

__global__ __launch_bounds__(256) void init_state_kernel(bf16_t* __restrict__ h0,
                                                         float* __restrict__ c, int n) {
    int i = blockIdx.x * blockDim.x + threadIdx.x;
    if (i < n) { h0[i] = (bf16_t)0.0f; c[i] = 0.0f; }
}

// ---------------- input projections: Gx[z] = sigmoid(X@Wm^T)*(X@W^T) + b_z ----------------
// grid: (H/64, BT/16, 3 gates {f,o,c}); block: 128 (4 waves, one 16-col chunk each)

__global__ __launch_bounds__(128) void input_proj_kernel(
    const bf16_t* __restrict__ Xbf, const bf16_t* __restrict__ Wbf,
    const float* __restrict__ b_f, const float* __restrict__ b_o, const float* __restrict__ b_c,
    bf16_t* __restrict__ Gx) {
    const int wave = threadIdx.x >> 5;
    const int lane = threadIdx.x & 31;
    const int n0 = blockIdx.x * 64 + wave * 16;
    const int m0 = blockIdx.y * 16;          // row in [BT]
    const int z  = blockIdx.z;               // 0=f 1=o 2=c
    const bf16_t* Wg = Wbf + (size_t)z * 2 * H_ * IN_;
    const bf16_t* Wm = Wg + (size_t)H_ * IN_;

    v8f accw = {}, accm = {};
    #pragma unroll 2
    for (int k0 = 0; k0 < IN_; k0 += 32) {
        v16bf a = load_a_frag(Xbf, IN_, m0, k0, lane);
        accw = wmma_bf16(a, load_b_frag(Wg, IN_, n0, k0, lane), accw);
        accm = wmma_bf16(a, load_b_frag(Wm, IN_, n0, k0, lane), accm);
    }

    const float* bias = (z == 0) ? b_f : ((z == 1) ? b_o : b_c);
    bf16_t* Gz = Gx + (size_t)z * BT_ * H_;
    #pragma unroll
    for (int v = 0; v < 8; ++v) {
        int m = m0 + v + ((lane & 16) ? 8 : 0);
        int n = n0 + (lane & 15);
        float val = sigm(accm[v]) * accw[v] + bias[n];
        Gz[(size_t)m * H_ + n] = (bf16_t)val;
    }
}

// ---------------- one recurrent step (f,o,c gates + elementwise update fused) ----------------
// grid: (H/64, B/32); block: 128 (4 waves). Each block: [32 x 64] tile of h_next.
// h rows staged once per block into LDS via async-to-LDS DMA; U fragments stream from L2.
// Ubf = [Uf,Ufm,Uo,Uom,Uc,Ucm] each H*H bf16.

__global__ __launch_bounds__(128) void lstm_step_kernel(
    const bf16_t* __restrict__ hprev, bf16_t* __restrict__ hnext,
    float* __restrict__ cbuf, const bf16_t* __restrict__ Ubf,
    const bf16_t* __restrict__ Gx, float* __restrict__ out, int t) {
    __shared__ bf16_t hs[32 * H_];           // 64 KB: h rows [mbase, mbase+32)

    const int tid   = threadIdx.x;
    const int wave  = tid >> 5;
    const int lane  = tid & 31;
    const int n0    = blockIdx.x * 64 + wave * 16;
    const int mbase = blockIdx.y * 32;

    // ---- stage h slab to LDS (ASYNCcnt-tracked DMA, no VGPR round-trip) ----
    {
        const bf16_t* src = hprev + (size_t)mbase * H_;   // 32768 contiguous bf16
        #pragma unroll
        for (int it = 0; it < 32; ++it) {
            int off = (it * 128 + tid) * 8;               // 8 bf16 = 16B per lane
            unsigned lds = (unsigned)(uintptr_t)(hs + off);
            const bf16_t* g = src + off;
            asm volatile("global_load_async_to_lds_b128 %0, %1, off"
                         :: "v"(lds), "v"(g) : "memory");
        }
        asm volatile("s_wait_asynccnt 0" ::: "memory");
    }
    __syncthreads();

    v8f acc[2][6];
    #pragma unroll
    for (int mt = 0; mt < 2; ++mt)
        #pragma unroll
        for (int i = 0; i < 6; ++i) acc[mt][i] = (v8f){};

    for (int k0 = 0; k0 < H_; k0 += 32) {
        v16bf a0 = load_a_frag(hs, H_, 0,  k0, lane);     // ds_load from LDS slab
        v16bf a1 = load_a_frag(hs, H_, 16, k0, lane);
        #pragma unroll
        for (int i = 0; i < 6; ++i) {
            v16bf b = load_b_frag(Ubf + (size_t)i * H_ * H_, H_, n0, k0, lane);
            acc[0][i] = wmma_bf16(a0, b, acc[0][i]);
            acc[1][i] = wmma_bf16(a1, b, acc[1][i]);
        }
    }

    const bf16_t* Gf = Gx;
    const bf16_t* Go = Gx + (size_t)BT_ * H_;
    const bf16_t* Gc = Gx + (size_t)2 * BT_ * H_;

    #pragma unroll
    for (int mt = 0; mt < 2; ++mt) {
        #pragma unroll
        for (int v = 0; v < 8; ++v) {
            int m = mbase + mt * 16 + v + ((lane & 16) ? 8 : 0);
            int n = n0 + (lane & 15);
            size_t oidx = ((size_t)m * T_ + t) * H_ + n;   // [B,T,H]
            size_t sidx = (size_t)m * H_ + n;              // [B,H]
            float fpre = sigm(acc[mt][1][v]) * acc[mt][0][v] + (float)Gf[oidx];
            float opre = sigm(acc[mt][3][v]) * acc[mt][2][v] + (float)Go[oidx];
            float cpre = sigm(acc[mt][5][v]) * acc[mt][4][v] + (float)Gc[oidx];
            float cold = cbuf[sidx];
            float cnew = sigm(fpre) * cold + tanh_fast(cpre);
            float hnew = sigm(opre) * cnew;
            cbuf[sidx]  = cnew;
            hnext[sidx] = (bf16_t)hnew;
            out[oidx]   = hnew;
        }
    }
}

// ---------------- finalize: hT = out[:,T-1,:], cT = cbuf ----------------

__global__ __launch_bounds__(256) void finalize_kernel(const float* __restrict__ out,
                                                       const float* __restrict__ cbuf,
                                                       float* __restrict__ hT,
                                                       float* __restrict__ cT) {
    int i = blockIdx.x * blockDim.x + threadIdx.x;
    if (i < B_ * H_) {
        int b = i / H_, n = i % H_;
        hT[i] = out[((size_t)b * T_ + (T_ - 1)) * H_ + n];
        cT[i] = cbuf[i];
    }
}

// ---------------- host launcher ----------------

extern "C" void kernel_launch(void* const* d_in, const int* in_sizes, int n_in,
                              void* d_out, int out_size, void* d_ws, size_t ws_size,
                              hipStream_t stream) {
    (void)in_sizes; (void)n_in; (void)out_size; (void)ws_size;
    // input order: input, then per gate (f,i,o,c): W, W_mask, U, U_mask, b
    const float* X   = (const float*)d_in[0];
    const float* bf_ = (const float*)d_in[5];
    const float* bo_ = (const float*)d_in[15];
    const float* bc_ = (const float*)d_in[20];
    // gate i (indices 6..10) is computed-but-unused in the reference -> skipped.
    const float* Wsrc[6] = { (const float*)d_in[1],  (const float*)d_in[2],
                             (const float*)d_in[11], (const float*)d_in[12],
                             (const float*)d_in[16], (const float*)d_in[17] };
    const float* Usrc[6] = { (const float*)d_in[3],  (const float*)d_in[4],
                             (const float*)d_in[13], (const float*)d_in[14],
                             (const float*)d_in[18], (const float*)d_in[19] };

    // workspace layout (bf16 elements unless noted)
    bf16_t* Xbf = (bf16_t*)d_ws;                          // BT*IN
    bf16_t* Wbf = Xbf + (size_t)BT_ * IN_;                // 6*H*IN
    bf16_t* Ubf = Wbf + (size_t)6 * H_ * IN_;             // 6*H*H
    bf16_t* Gx  = Ubf + (size_t)6 * H_ * H_;              // 3*BT*H
    bf16_t* h0  = Gx  + (size_t)3 * BT_ * H_;             // B*H
    bf16_t* h1  = h0  + (size_t)B_ * H_;                  // B*H
    float*  cbuf = (float*)(h1 + (size_t)B_ * H_);        // B*H f32

    float* out = (float*)d_out;
    float* hT  = out + (size_t)BT_ * H_;
    float* cT  = hT + (size_t)B_ * H_;

    // 1) convert input + weights to bf16
    {
        int n = BT_ * IN_;
        f32_to_bf16_kernel<<<n / 1024, 256, 0, stream>>>(X, Xbf, n);
    }
    for (int i = 0; i < 6; ++i) {
        int nw = H_ * IN_;
        f32_to_bf16_kernel<<<nw / 1024, 256, 0, stream>>>(Wsrc[i], Wbf + (size_t)i * nw, nw);
        int nu = H_ * H_;
        f32_to_bf16_kernel<<<nu / 1024, 256, 0, stream>>>(Usrc[i], Ubf + (size_t)i * nu, nu);
    }

    // 2) zero initial state
    init_state_kernel<<<(B_ * H_) / 256, 256, 0, stream>>>(h0, cbuf, B_ * H_);

    // 3) hoisted input projections with fused mask-sigmoid + bias
    input_proj_kernel<<<dim3(H_ / 64, BT_ / 16, 3), 128, 0, stream>>>(
        Xbf, Wbf, bf_, bo_, bc_, Gx);

    // 4) sequential recurrence, one fused kernel per timestep, h ping-pong
    for (int t = 0; t < T_; ++t) {
        const bf16_t* hp = (t & 1) ? h1 : h0;
        bf16_t*       hn = (t & 1) ? h0 : h1;
        lstm_step_kernel<<<dim3(H_ / 64, B_ / 32), 128, 0, stream>>>(
            hp, hn, cbuf, Ubf, Gx, out, t);
    }

    // 5) final states
    finalize_kernel<<<(B_ * H_) / 256, 256, 0, stream>>>(out, cbuf, hT, cT);
}
